// IRNNBase_82411832475654
// MI455X (gfx1250) — matrix operand order repaired and checked by reading, not compiled, not run on hardware
//
#include <hip/hip_runtime.h>

// ---------------- CDNA5 WMMA types ----------------
typedef __attribute__((ext_vector_type(16))) __bf16 v16bf;
typedef __attribute__((ext_vector_type(8)))  float  v8f;
typedef __attribute__((ext_vector_type(8)))  __bf16 bf16x8;

#define S_LEN 2048
#define BATCH 64
#define IDIM  256
#define HDIM  256
#define GDIM  1024   // 4*H
#define KDIM  512    // I + H

// LDS strides (bf16 elements). 264*2B = 528B pitch: 16B aligned (b128 legal)
// and 528/4 = 132 banks -> rows/cols land 4 banks apart -> conflict-free
// half-wave b128 accesses.
#define XSTR 264
#define GSTR 1040     // fused-path gate buffer pitch
#define WCOLS 480     // W_h column-tiles 0..29 cached in LDS (253,440 B)

// Workspace layout
#define WS_WT_OFF   ((size_t)0)
#define WS_BIAS_OFF ((size_t)GDIM * KDIM * 2)            // 1 MB
#define WS_XG_OFF   (WS_BIAS_OFF + (size_t)GDIM * 4)     // +4 KB
#define XG_BYTES    ((size_t)S_LEN * 256 * 512)          // 256 MB (frag-layout bf16)

// --------------------------------------------------------------------------
// Paired-tile K=256 reduction (xproj / fused path): shared A, two B streams.
// --------------------------------------------------------------------------
__device__ __forceinline__ void gemm_k256_x2(const __bf16* __restrict__ arow,
                                             const __bf16* __restrict__ w0,
                                             const __bf16* __restrict__ w1,
                                             int half, v8f& acc0, v8f& acc1)
{
    #pragma unroll 4
    for (int kb = 0; kb < 8; ++kb) {
        const int kbase = kb << 5;
        const __bf16* ap = arow + kbase + (half << 4);
        const bf16x8 a0 = *(const bf16x8*)(ap);
        const bf16x8 a1 = *(const bf16x8*)(ap + 8);
        const __bf16* bp0 = w0 + kbase + (half << 4);
        const __bf16* bp1 = w1 + kbase + (half << 4);
        const bf16x8 b00 = *(const bf16x8*)(bp0);
        const bf16x8 b01 = *(const bf16x8*)(bp0 + 8);
        const bf16x8 b10 = *(const bf16x8*)(bp1);
        const bf16x8 b11 = *(const bf16x8*)(bp1 + 8);
        v16bf af, bf0, bf1;
        #pragma unroll
        for (int e = 0; e < 8; ++e) {
            af[e]  = a0[e];  af[e + 8]  = a1[e];
            bf0[e] = b00[e]; bf0[e + 8] = b01[e];
            bf1[e] = b10[e]; bf1[e + 8] = b11[e];
        }
        acc0 = __builtin_amdgcn_wmma_f32_16x16x32_bf16(false, af, false, bf0,
                                                       (short)0, acc0, false, false);
        acc1 = __builtin_amdgcn_wmma_f32_16x16x32_bf16(false, af, false, bf1,
                                                       (short)0, acc1, false, false);
    }
}

// --------------------------------------------------------------------------
// Quad-stream K=256 reduction (recurrence): one shared A stream, four B
// streams (i/f/g/o quarters; LDS or L2 per pointer), four WMMA chains.
// --------------------------------------------------------------------------
__device__ __forceinline__ void gemm_k256_x4(const __bf16* arow,
                                             const __bf16* w0, const __bf16* w1,
                                             const __bf16* w2, const __bf16* w3,
                                             int half,
                                             v8f& A0, v8f& A1, v8f& A2, v8f& A3)
{
    #pragma unroll 2
    for (int kb = 0; kb < 8; ++kb) {
        const int kbase = kb << 5;
        const __bf16* ap = arow + kbase + (half << 4);
        const bf16x8 a0 = *(const bf16x8*)(ap);
        const bf16x8 a1 = *(const bf16x8*)(ap + 8);
        v16bf af;
        #pragma unroll
        for (int e = 0; e < 8; ++e) { af[e] = a0[e]; af[e + 8] = a1[e]; }

        const __bf16* bp[4] = { w0 + kbase + (half << 4), w1 + kbase + (half << 4),
                                w2 + kbase + (half << 4), w3 + kbase + (half << 4) };
        v8f* accs[4] = { &A0, &A1, &A2, &A3 };
        #pragma unroll
        for (int q = 0; q < 4; ++q) {
            const bf16x8 b0 = *(const bf16x8*)(bp[q]);
            const bf16x8 b1 = *(const bf16x8*)(bp[q] + 8);
            v16bf bf;
            #pragma unroll
            for (int e = 0; e < 8; ++e) { bf[e] = b0[e]; bf[e + 8] = b1[e]; }
            *accs[q] = __builtin_amdgcn_wmma_f32_16x16x32_bf16(
                           false, af, false, bf, (short)0, *accs[q], false, false);
        }
    }
}

// Fused-path LSTM cell (LDS gate buffer variant).
__device__ __forceinline__ void lstm_cell(const __bf16* gls, __bf16* hls,
                                          float* __restrict__ out,
                                          float creg[16], int s, int eb, int ej)
{
    const size_t O_hN = (size_t)S_LEN * BATCH * HDIM;
    const size_t O_c  = O_hN + (size_t)BATCH * HDIM;
    const size_t O_cN = O_c + (size_t)S_LEN * BATCH * HDIM;

    const __bf16* gr = gls + eb * GSTR;
    float* hdst = out + (size_t)s * BATCH * HDIM + (size_t)eb * HDIM + ej;
    float* cdst = out + O_c + (size_t)s * BATCH * HDIM + (size_t)eb * HDIM + ej;
    float hn[16];
    #pragma unroll
    for (int u = 0; u < 16; ++u) {
        const float gi = (float)gr[      ej + u];
        const float gf = (float)gr[256 + ej + u];
        const float gg = (float)gr[512 + ej + u];
        const float go = (float)gr[768 + ej + u];
        const float si = 1.f / (1.f + __expf(-gi));
        const float sf = 1.f / (1.f + __expf(-gf));
        const float tg = 2.f / (1.f + __expf(-2.f * gg)) - 1.f;
        const float so = 1.f / (1.f + __expf(-go));
        const float cn = sf * creg[u] + si * tg;
        const float th = 2.f / (1.f + __expf(-2.f * cn)) - 1.f;
        const float h  = so * th;
        creg[u] = cn;  hn[u] = h;  cdst[u] = cn;  hdst[u] = h;
    }
    #pragma unroll
    for (int u = 0; u < 16; ++u)
        hls[eb * XSTR + ej + u] = (__bf16)hn[u];
    if (s == S_LEN - 1) {
        #pragma unroll
        for (int u = 0; u < 16; ++u) {
            out[O_hN + (size_t)eb * HDIM + ej + u] = hn[u];
            out[O_cN + (size_t)eb * HDIM + ej + u] = creg[u];
        }
    }
}

// --------------------------------------------------------------------------
// Kernel 0: pack [W_x ; W_h] -> bf16 N-major WT[g][k]; bias = bx + bh.
// --------------------------------------------------------------------------
__global__ void lstm_pack_weights(const float* __restrict__ w_x,
                                  const float* __restrict__ w_h,
                                  const float* __restrict__ bx,
                                  const float* __restrict__ bh,
                                  __bf16* __restrict__ WT,
                                  float* __restrict__ biasF)
{
    int idx = blockIdx.x * blockDim.x + threadIdx.x;
    if (idx < GDIM * KDIM) {
        int g = idx >> 9;
        int k = idx & 511;
        float v = (k < IDIM) ? w_x[(size_t)k * GDIM + g]
                             : w_h[(size_t)(k - IDIM) * GDIM + g];
        WT[(size_t)g * KDIM + k] = (__bf16)v;
    }
    if (idx < GDIM) biasF[idx] = bx[idx] + bh[idx];
}

// --------------------------------------------------------------------------
// Kernel 1a: grid-parallel xg = x @ W_x + bias; bf16 C-fragment output,
//   elem offset = ((s*256 + tile)*32 + lane)*8,  tile = rt*64 + ct
// --------------------------------------------------------------------------
__global__ __launch_bounds__(1024)
void lstm_xproj(const float* __restrict__ x, const __bf16* __restrict__ WT,
                const float* __restrict__ biasF, __bf16* __restrict__ xg)
{
    __shared__ __bf16 xls[BATCH * XSTR];
    const int s    = blockIdx.x;
    const int t    = threadIdx.x;
    const int lane = t & 31;
    const int wv   = t >> 5;
    const int half = lane >> 4;
    const int ln   = lane & 15;
    const int eb   = t >> 4;
    const int ej   = (t & 15) << 4;

    const int rt   = wv >> 3;
    const int ctb  = wv & 7;
    const int arow = (rt << 4) + ln;

    const float* srcf = x + (size_t)s * BATCH * IDIM + (size_t)eb * IDIM + ej;
    #pragma unroll
    for (int u = 0; u < 16; ++u)
        xls[eb * XSTR + ej + u] = (__bf16)srcf[u];
    __syncthreads();

    for (int p = 0; p < 4; ++p) {
        const int ct0 = ctb + (p << 4);
        const int ct1 = ct0 + 8;
        const int bcol0 = (ct0 << 4) + ln;
        const int bcol1 = (ct1 << 4) + ln;

        v8f acc0, acc1;
        const float bv0 = biasF[bcol0];
        const float bv1 = biasF[bcol1];
        #pragma unroll
        for (int e = 0; e < 8; ++e) { acc0[e] = bv0; acc1[e] = bv1; }

        gemm_k256_x2(&xls[arow * XSTR],
                     WT + (size_t)bcol0 * KDIM, WT + (size_t)bcol1 * KDIM,
                     half, acc0, acc1);

        bf16x8 p0, p1;
        #pragma unroll
        for (int e = 0; e < 8; ++e) { p0[e] = (__bf16)acc0[e]; p1[e] = (__bf16)acc1[e]; }
        const size_t tile0 = (size_t)(rt << 6) + ct0;
        const size_t tile1 = (size_t)(rt << 6) + ct1;
        *(bf16x8*)(xg + ((((size_t)s << 8) + tile0) << 8) + (lane << 3)) = p0;
        *(bf16x8*)(xg + ((((size_t)s << 8) + tile1) << 8) + (lane << 3)) = p1;
    }
}

// --------------------------------------------------------------------------
// Kernel 2a: persistent recurrence, K=256, in-register cell update.
// Wave owns row-tile rt and h-column-tiles jt0, jt0+8; for each jt it
// computes all four gate quarters (ct = jt, jt+16, jt+32, jt+48) so the
// cell runs on WMMA fragments directly (no gate LDS buffer). W_h tiles
// 0..29 live in LDS; h is ping-pong double-buffered -> 1 barrier/step.
// --------------------------------------------------------------------------
__device__ __forceinline__ void rec_step(int s, const __bf16* hcur, __bf16* hnxt,
                                         const __bf16* __restrict__ WT,
                                         const __bf16* wls,
                                         const __bf16* __restrict__ xg,
                                         float* __restrict__ out,
                                         float cfrag[2][8],
                                         int rt, int jt0, int half, int ln, int lane)
{
    const size_t O_hN = (size_t)S_LEN * BATCH * HDIM;
    const size_t O_c  = O_hN + (size_t)BATCH * HDIM;
    const size_t O_cN = O_c + (size_t)S_LEN * BATCH * HDIM;

    const __bf16* ar = hcur + (size_t)((rt << 4) + ln) * XSTR;

    #pragma unroll
    for (int j = 0; j < 2; ++j) {
        const int jt = jt0 + (j << 3);           // 0..15

        // Seed accumulators from xg fragments (quarter tiles are +16 apart).
        const size_t fb = ((((size_t)s << 8) + (size_t)(rt << 6) + jt) << 8) + (lane << 3);
        const bf16x8 qi = *(const bf16x8*)(xg + fb);
        const bf16x8 qf = *(const bf16x8*)(xg + fb + (16 << 8));
        const bf16x8 qg = *(const bf16x8*)(xg + fb + (32 << 8));
        const bf16x8 qo = *(const bf16x8*)(xg + fb + (48 << 8));
        if (s + 1 < S_LEN) {
            __builtin_prefetch(xg + fb + ((size_t)256 << 8), 0, 0);
            __builtin_prefetch(xg + fb + ((size_t)(256 + 32) << 8), 0, 0);
        }
        v8f ai, afr, ag, ao;
        #pragma unroll
        for (int e = 0; e < 8; ++e) {
            ai[e] = (float)qi[e]; afr[e] = (float)qf[e];
            ag[e] = (float)qg[e]; ao[e]  = (float)qo[e];
        }

        // B streams: q0,q1 from LDS W-cache when tile < 30, else L2.
        const __bf16* b0 = wls + (size_t)((jt << 4) + ln) * XSTR;
        const __bf16* b2 = WT + (size_t)(((jt + 32) << 4) + ln) * KDIM + IDIM;
        const __bf16* b3 = WT + (size_t)(((jt + 48) << 4) + ln) * KDIM + IDIM;
        if (jt < 14) {
            const __bf16* b1 = wls + (size_t)(((jt + 16) << 4) + ln) * XSTR;
            gemm_k256_x4(ar, b0, b1, b2, b3, half, ai, afr, ag, ao);
        } else {
            const __bf16* b1 = WT + (size_t)(((jt + 16) << 4) + ln) * KDIM + IDIM;
            gemm_k256_x4(ar, b0, b1, b2, b3, half, ai, afr, ag, ao);
        }

        // Cell update directly on fragments: elem e -> (row 16rt+8*half+e,
        // col 16jt+ln); identical (M,N) across the four quarters.
        #pragma unroll
        for (int e = 0; e < 8; ++e) {
            const float si = 1.f / (1.f + __expf(-ai[e]));
            const float sf = 1.f / (1.f + __expf(-afr[e]));
            const float tg = 2.f / (1.f + __expf(-2.f * ag[e])) - 1.f;
            const float so = 1.f / (1.f + __expf(-ao[e]));
            const float cn = sf * cfrag[j][e] + si * tg;
            const float h  = so * (2.f / (1.f + __expf(-2.f * cn)) - 1.f);
            cfrag[j][e] = cn;

            const int row = (rt << 4) + (half << 3) + e;
            const int col = (jt << 4) + ln;
            const size_t idx = (size_t)s * BATCH * HDIM + (size_t)row * HDIM + col;
            out[idx]       = h;
            out[O_c + idx] = cn;
            hnxt[row * XSTR + col] = (__bf16)h;
            if (s == S_LEN - 1) {
                out[O_hN + (size_t)row * HDIM + col] = h;
                out[O_cN + (size_t)row * HDIM + col] = cn;
            }
        }
    }
    __syncthreads();   // h_s visible to all waves of this rt before step s+1
}

__global__ __launch_bounds__(1024)
void lstm_rec_xg(const float* __restrict__ h0, const float* __restrict__ c0,
                 const __bf16* __restrict__ WT, const __bf16* __restrict__ xg,
                 float* __restrict__ out)
{
    __shared__ __bf16 hlsA[BATCH * XSTR];     // 33,792 B
    __shared__ __bf16 hlsB[BATCH * XSTR];     // 33,792 B
    __shared__ __bf16 wls[WCOLS * XSTR];      // 253,440 B : W_h cols 0..479

    const int t    = threadIdx.x;
    const int lane = t & 31;
    const int wv   = t >> 5;
    const int half = lane >> 4;
    const int ln   = lane & 15;
    const int eb   = t >> 4;
    const int ej   = (t & 15) << 4;

    const int rt  = wv >> 3;      // 0..3
    const int jt0 = wv & 7;       // jt in {jt0, jt0+8}

    // Stage W_h LDS cache (cols 0..479, N-major, padded stride).
    if (t < 960) {
        const int col = t >> 1;
        const int ko  = (t & 1) << 7;
        const __bf16* src = WT + (size_t)col * KDIM + IDIM + ko;
        __bf16* dst = wls + (size_t)col * XSTR + ko;
        #pragma unroll
        for (int u = 0; u < 16; ++u)
            *(bf16x8*)(dst + (u << 3)) = *(const bf16x8*)(src + (u << 3));
    }
    // h0 -> read buffer for step 0; c0 -> fragment registers.
    #pragma unroll
    for (int u = 0; u < 16; ++u)
        hlsA[eb * XSTR + ej + u] = (__bf16)h0[eb * HDIM + ej + u];
    float cfrag[2][8];
    #pragma unroll
    for (int j = 0; j < 2; ++j) {
        const int jt = jt0 + (j << 3);
        #pragma unroll
        for (int e = 0; e < 8; ++e)
            cfrag[j][e] = c0[(size_t)((rt << 4) + (half << 3) + e) * HDIM + (jt << 4) + ln];
    }
    __syncthreads();

    for (int s2 = 0; s2 < S_LEN; s2 += 2) {
        rec_step(s2,     hlsA, hlsB, WT, wls, xg, out, cfrag, rt, jt0, half, ln, lane);
        rec_step(s2 + 1, hlsB, hlsA, WT, wls, xg, out, cfrag, rt, jt0, half, ln, lane);
    }
}

// --------------------------------------------------------------------------
// Kernel 1b (small-ws fallback): fully fused persistent LSTM, K=512.
// --------------------------------------------------------------------------
__global__ __launch_bounds__(1024)
void lstm_fused_persistent(const float* __restrict__ x,
                           const float* __restrict__ h0,
                           const float* __restrict__ c0,
                           const __bf16* __restrict__ WT,
                           const float* __restrict__ biasF,
                           float* __restrict__ out)
{
    __shared__ __bf16 xls[BATCH * XSTR];
    __shared__ __bf16 hls[BATCH * XSTR];
    __shared__ __bf16 gls[BATCH * GSTR];

    const int t    = threadIdx.x;
    const int lane = t & 31;
    const int wv   = t >> 5;
    const int half = lane >> 4;
    const int ln   = lane & 15;
    const int eb   = t >> 4;
    const int ej   = (t & 15) << 4;

    const int rt   = wv >> 3;
    const int ctb  = wv & 7;
    const int arow = (rt << 4) + ln;

    float creg[16];
    #pragma unroll
    for (int u = 0; u < 16; ++u) creg[u] = c0[eb * HDIM + ej + u];
    #pragma unroll
    for (int u = 0; u < 16; ++u)
        hls[eb * XSTR + ej + u] = (__bf16)h0[eb * HDIM + ej + u];
    __syncthreads();

    for (int s = 0; s < S_LEN; ++s) {
        {
            const float* srcf = x + (size_t)s * BATCH * IDIM + (size_t)eb * IDIM + ej;
            #pragma unroll
            for (int u = 0; u < 16; ++u)
                xls[eb * XSTR + ej + u] = (__bf16)srcf[u];
            if (s + 1 < S_LEN)
                __builtin_prefetch(x + (size_t)(s + 1) * BATCH * IDIM + (size_t)eb * IDIM + ej, 0, 0);
        }
        __syncthreads();

        for (int p = 0; p < 4; ++p) {
            const int ct0 = ctb + (p << 4);
            const int ct1 = ct0 + 8;
            const int bcol0 = (ct0 << 4) + ln;
            const int bcol1 = (ct1 << 4) + ln;

            v8f acc0, acc1;
            const float bv0 = biasF[bcol0];
            const float bv1 = biasF[bcol1];
            #pragma unroll
            for (int e = 0; e < 8; ++e) { acc0[e] = bv0; acc1[e] = bv1; }

            const __bf16* w0 = WT + (size_t)bcol0 * KDIM;
            const __bf16* w1 = WT + (size_t)bcol1 * KDIM;
            gemm_k256_x2(&xls[arow * XSTR], w0, w1, half, acc0, acc1);
            gemm_k256_x2(&hls[arow * XSTR], w0 + IDIM, w1 + IDIM, half, acc0, acc1);

            #pragma unroll
            for (int e = 0; e < 8; ++e) {
                const int gm = (rt << 4) + (half << 3) + e;
                gls[gm * GSTR + (ct0 << 4) + ln] = (__bf16)acc0[e];
                gls[gm * GSTR + (ct1 << 4) + ln] = (__bf16)acc1[e];
            }
        }
        __syncthreads();
        lstm_cell(gls, hls, out, creg, s, eb, ej);
        __syncthreads();
    }
}

// --------------------------------------------------------------------------
extern "C" void kernel_launch(void* const* d_in, const int* in_sizes, int n_in,
                              void* d_out, int out_size, void* d_ws, size_t ws_size,
                              hipStream_t stream) {
    const float* x   = (const float*)d_in[0];
    const float* w_x = (const float*)d_in[1];
    const float* w_h = (const float*)d_in[2];
    const float* bx  = (const float*)d_in[3];
    const float* bh  = (const float*)d_in[4];
    const float* h0  = (const float*)d_in[5];
    const float* c0  = (const float*)d_in[6];

    __bf16* WT    = (__bf16*)((char*)d_ws + WS_WT_OFF);
    float*  biasF = (float*)((char*)d_ws + WS_BIAS_OFF);
    __bf16* xg    = (__bf16*)((char*)d_ws + WS_XG_OFF);

    lstm_pack_weights<<<(GDIM * KDIM + 255) / 256, 256, 0, stream>>>(
        w_x, w_h, bx, bh, WT, biasF);

    if (ws_size >= WS_XG_OFF + XG_BYTES) {
        lstm_xproj<<<S_LEN, 1024, 0, stream>>>(x, WT, biasF, xg);
        lstm_rec_xg<<<1, 1024, 0, stream>>>(h0, c0, WT, xg, (float*)d_out);
    } else {
        lstm_fused_persistent<<<1, 1024, 0, stream>>>(
            x, h0, c0, WT, biasF, (float*)d_out);
    }
}
